// SubModule_32804960207287
// MI455X (gfx1250) — matrix operand order repaired and checked by reading, compiled, not run on hardware
//
#include <hip/hip_runtime.h>

// WaveNet stack for MI455X (gfx1250, wave32, WMMA).
// Per layer: conv1x1(+relu) -> dilated k=2 convs (f,g) -> tanh*sigmoid ->
// conv1x1(+relu) -> residual + skip. All channel GEMMs run on
// v_wmma_f32_16x16x32_f16 (f16 in, f32 accumulate). Intermediates (h, x)
// live in d_ws (2 x 33.5 MB, resident in the 192 MB L2). Skip accumulates
// in d_out. 67,108,864 bytes of d_ws required.

typedef __attribute__((ext_vector_type(16))) _Float16 v16h;
typedef __attribute__((ext_vector_type(8)))  float    v8f;

#define T_LEN 16384
#define NB 32
#define C1 16
#define C2 32
#define WAVES_PER_BLOCK 8
#define TILES_PER_WAVE 4
#define BLOCK_THREADS 256
// tiles of 16 positions: NB*T_LEN/16 = 32768 ; blocks = 32768/(8*4) = 1024
#define GRID_BLOCKS 1024

static __device__ __forceinline__ float fast_rcp(float x) {
  return __builtin_amdgcn_rcpf(x);  // single v_rcp_f32 (TRANS), no IEEE refine
}
// tanh(x) = 1 - 2/(exp(2x)+1); rcp(inf)=0 so saturation at +/-1 is exact.
static __device__ __forceinline__ float fast_tanh(float x) {
  return 1.f - 2.f * fast_rcp(__expf(2.f * x) + 1.f);
}
// sigmoid(x) = 1/(1+exp(-x)); rcp(inf)=0 handles the -inf side.
static __device__ __forceinline__ float fast_sigmoid(float x) {
  return fast_rcp(1.f + __expf(-x));
}
static __device__ __forceinline__ float relu_f(float x) { return x > 0.f ? x : 0.f; }

// ---------------------------------------------------------------------------
// Kernel A: h = relu(W1 @ x + b1), pointwise over T. One 16x16x32 WMMA per
// 16-position tile (K zero-padded from KIN to 32).
// x: (NB, KIN, T)  w: (16, KIN)  bias: (16)  h: (NB, 16, T)
// ---------------------------------------------------------------------------
template <int KIN>
__global__ __launch_bounds__(BLOCK_THREADS) void conv1_relu_kernel(
    const float* __restrict__ xin, const float* __restrict__ w,
    const float* __restrict__ bias, float* __restrict__ h) {
  const int lane  = threadIdx.x & 31;
  const int wave  = threadIdx.x >> 5;
  const int m     = lane & 15;           // A-frag row / C-frag column
  const int khalf = (lane & 16) >> 1;    // A-frag K-half select (0 or 8)
  const int kbase = lane & 16;           // unused here; B-frag uses khalf order like A-frag
  const int m0    = (lane & 16) >> 1;    // C-frag row offset (0 or 8)

  // A fragment (weights, constant across tiles). wave32 16-bit A layout:
  // lanes<16: e0..7->K=0..7, e8..15->K=16..23 ; lanes>=16: +8.
  v16h a;
#pragma unroll
  for (int e = 0; e < 16; ++e) {
    int k = khalf + e + (e & 8);
    a[e] = (k < KIN) ? (_Float16)w[m * KIN + k] : (_Float16)0.f;
  }
  const float bias0 = bias[m0];  // hoist per-lane bias loads
  float bvals[8];
#pragma unroll
  for (int r = 0; r < 8; ++r) bvals[r] = bias[m0 + r];
  (void)bias0;

  const int tile0 = (blockIdx.x * WAVES_PER_BLOCK + wave) * TILES_PER_WAVE;
  for (int j = 0; j < TILES_PER_WAVE; ++j) {
    const int tile = tile0 + j;
    const int b    = tile >> 10;              // T/16 = 1024 tiles per batch row
    const int t0   = (tile & 1023) << 4;
    const int n    = lane & 15;
    const int t    = t0 + n;

    // B fragment: B[k][n] = x[b,k,t], K zero-padded to 32. 32-bit offsets.
    const int xb = (b * KIN) * T_LEN + t;
    v16h bvec;
#pragma unroll
    for (int e = 0; e < 16; ++e) {
      int k   = khalf + e + (e & 8);
      bvec[e] = (k < KIN) ? (_Float16)xin[xb + k * T_LEN] : (_Float16)0.f;
    }

    v8f c = {};
    c = __builtin_amdgcn_wmma_f32_16x16x32_f16(false, a, false, bvec, (short)0,
                                               c, false, false);

    const int base = (b * C1 + m0) * T_LEN + t;
#pragma unroll
    for (int r = 0; r < 8; ++r) {
      h[base + r * T_LEN] = relu_f(c[r] + bvals[r]);
    }
  }
}

// ---------------------------------------------------------------------------
// Kernel B: gated residual unit.
//   f = Wf @ [h(t-dl); h(t+dr)] + bf   (two 16x16x32 WMMAs, K = 2 taps x 16ch)
//   g = Wg @ [...] + bg                (two WMMAs)
//   gated = tanh(f)*sigmoid(g)         (TRANS/VALU, co-executes with XDL)
//   z = relu(W2 @ gated + b2)          (one WMMA; gated repacked via LDS)
//   x_out = h + z ; skip (+)= z ; last layer: out = relu(skip + z)
// mode: 0 = first (skip = z), 1 = mid (skip += z), 2 = last (relu into skip)
// ---------------------------------------------------------------------------
__global__ __launch_bounds__(BLOCK_THREADS) void gated_res_kernel(
    const float* __restrict__ h, const float* __restrict__ wf,
    const float* __restrict__ bfv, const float* __restrict__ wg,
    const float* __restrict__ bgv, const float* __restrict__ w2,
    const float* __restrict__ b2v, float* __restrict__ xout,
    float* __restrict__ skip, int dil, int mode) {
  __shared__ float lds_g[WAVES_PER_BLOCK][C2 * 16];  // 16 KB, per-wave slabs

  const int lane  = threadIdx.x & 31;
  const int wave  = threadIdx.x >> 5;
  const int m     = lane & 15;
  const int khalf = (lane & 16) >> 1;
  const int kbase = lane & 16;
  const int m0    = (lane & 16) >> 1;
  const int dl    = dil >> 1;
  const int dr    = dil - dl;

  // A fragments (constant per layer). Stacked weights: K<16 -> tap0, K>=16 -> tap1.
  v16h af0, af1, ag0, ag1, a2;
#pragma unroll
  for (int e = 0; e < 16; ++e) {
    int k = kbase + e;  // 0..31
    int kc = k & 15, tap = k >> 4;
    af0[e] = (_Float16)wf[((0 + m) * 16 + kc) * 2 + tap];
    af1[e] = (_Float16)wf[((16 + m) * 16 + kc) * 2 + tap];
    ag0[e] = (_Float16)wg[((0 + m) * 16 + kc) * 2 + tap];
    ag1[e] = (_Float16)wg[((16 + m) * 16 + kc) * 2 + tap];
    a2[e]  = (_Float16)w2[m * C2 + k];
  }
  // Hoist bias loads (per-lane constants across tiles).
  float bf0[8], bf1[8], bg0[8], bg1[8], b2r[8];
#pragma unroll
  for (int r = 0; r < 8; ++r) {
    bf0[r] = bfv[m0 + r];
    bf1[r] = bfv[16 + m0 + r];
    bg0[r] = bgv[m0 + r];
    bg1[r] = bgv[16 + m0 + r];
    b2r[r] = b2v[m0 + r];
  }

  const int tile0 = (blockIdx.x * WAVES_PER_BLOCK + wave) * TILES_PER_WAVE;
  for (int j = 0; j < TILES_PER_WAVE; ++j) {
    const int tile = tile0 + j;
    const int b    = tile >> 10;
    const int t0   = (tile & 1023) << 4;
    const int n    = lane & 15;

    // B fragment: rows K<16 = h[:, t-dl], rows K>=16 = h[:, t+dr] (zero pad OOB).
    const int tap = kbase >> 4;  // lanes<16 -> tap0, lanes>=16 -> tap1
    const int tt  = t0 + n + (tap ? dr : -dl);
    const bool inb = (tt >= 0) && (tt < T_LEN);
    const int ttc = tt < 0 ? 0 : (tt >= T_LEN ? T_LEN - 1 : tt);
    const int hb = b * C1 * T_LEN + ttc;
    v16h bfrag;
#pragma unroll
    for (int e = 0; e < 16; ++e) {
      float v  = h[hb + e * T_LEN];
      bfrag[e] = (_Float16)(inb ? v : 0.f);
    }

    v8f cf0 = {}, cf1 = {}, cg0 = {}, cg1 = {};
    cf0 = __builtin_amdgcn_wmma_f32_16x16x32_f16(false, af0, false, bfrag,
                                                 (short)0, cf0, false, false);
    cf1 = __builtin_amdgcn_wmma_f32_16x16x32_f16(false, af1, false, bfrag,
                                                 (short)0, cf1, false, false);
    cg0 = __builtin_amdgcn_wmma_f32_16x16x32_f16(false, ag0, false, bfrag,
                                                 (short)0, cg0, false, false);
    cg1 = __builtin_amdgcn_wmma_f32_16x16x32_f16(false, ag1, false, bfrag,
                                                 (short)0, cg1, false, false);

    // Gate on TRANS/VALU; stage gated[ch][n] in per-wave LDS for C->B relayout.
    float* g = lds_g[wave];
#pragma unroll
    for (int r = 0; r < 8; ++r) {
      int ch = m0 + r;
      g[ch * 16 + n] =
          fast_tanh(cf0[r] + bf0[r]) * fast_sigmoid(cg0[r] + bg0[r]);
      g[(16 + ch) * 16 + n] =
          fast_tanh(cf1[r] + bf1[r]) * fast_sigmoid(cg1[r] + bg1[r]);
    }
    __builtin_amdgcn_wave_barrier();  // wave-lockstep LDS exchange (no xwave)

    // Repack gated into B-fragment layout (K = channel 0..31).
    v16h b2f;
#pragma unroll
    for (int e = 0; e < 16; ++e) b2f[e] = (_Float16)g[(kbase + e) * 16 + n];

    v8f cz = {};
    cz = __builtin_amdgcn_wmma_f32_16x16x32_f16(false, a2, false, b2f, (short)0,
                                                cz, false, false);

    const int obase = (b * C1 + m0) * T_LEN + t0 + n;
#pragma unroll
    for (int r = 0; r < 8; ++r) {
      const int idx = obase + r * T_LEN;
      float z = relu_f(cz[r] + b2r[r]);
      if (mode == 2) {
        skip[idx] = relu_f(skip[idx] + z);  // final output in place
      } else {
        xout[idx] = h[idx] + z;  // residual: x_next = h + z
        if (mode == 0)
          skip[idx] = z;  // initialize (no memset needed)
        else
          skip[idx] += z;
      }
    }
    __builtin_amdgcn_wave_barrier();
  }
}

// ---------------------------------------------------------------------------
extern "C" void kernel_launch(void* const* d_in, const int* in_sizes, int n_in,
                              void* d_out, int out_size, void* d_ws,
                              size_t ws_size, hipStream_t stream) {
  const float* x        = (const float*)d_in[0];   // (32, 1, 16384)
  const float* w1_first = (const float*)d_in[1];   // (16, 1)
  const float* b1_first = (const float*)d_in[2];   // (16,)
  const float* w1_post  = (const float*)d_in[3];   // (32, 16, 16)
  const float* b1_post  = (const float*)d_in[4];   // (32, 16)
  const float* wf       = (const float*)d_in[5];   // (33, 32, 16, 2)
  const float* bfv      = (const float*)d_in[6];   // (33, 32)
  const float* wg       = (const float*)d_in[7];
  const float* bgv      = (const float*)d_in[8];
  const float* w2       = (const float*)d_in[9];   // (33, 16, 32)
  const float* b2v      = (const float*)d_in[10];  // (33, 16)
  float* out = (float*)d_out;                      // (32, 16, 16384) = skip acc

  const size_t plane = (size_t)NB * C1 * T_LEN;    // 8,388,608 floats
  float* h    = (float*)d_ws;                      // conv1 output
  float* xbuf = h + plane;                         // residual stream

  dim3 grid(GRID_BLOCKS), blk(BLOCK_THREADS);
  for (int i = 0; i < 33; ++i) {
    const int d = 1 << (i % 11);
    if (i == 0)
      conv1_relu_kernel<1><<<grid, blk, 0, stream>>>(x, w1_first, b1_first, h);
    else
      conv1_relu_kernel<16><<<grid, blk, 0, stream>>>(
          xbuf, w1_post + (size_t)(i - 1) * 256, b1_post + (size_t)(i - 1) * 16,
          h);
    const int mode = (i == 0) ? 0 : (i == 32 ? 2 : 1);
    gated_res_kernel<<<grid, blk, 0, stream>>>(
        h, wf + (size_t)i * C2 * C1 * 2, bfv + (size_t)i * C2,
        wg + (size_t)i * C2 * C1 * 2, bgv + (size_t)i * C2,
        w2 + (size_t)i * C1 * C2, b2v + (size_t)i * C1, xbuf, out, d, mode);
  }
}
